// vit_90744069030605
// MI455X (gfx1250) — compile-verified, matrix operand order checked by reading, not run on hardware
//
#include <hip/hip_runtime.h>
#include <hip/hip_bf16.h>

// ---------------------------------------------------------------------------
// Reference returns `inp` = relu(input + relu(bn2(conv2(relu(bn1(conv1(input)))))))
// in NHWC (B, N, C). Attention/MLP are dead code.
//
//   K1: conv1(1->256,3x3)+BN1+ReLU -> f16 activation, NHWC [B,X,Y,C]
//   K2: pack conv2 weights (BN2 scale folded) into per-lane WMMA-B order (f16)
//   K3: implicit-GEMM conv2 via v_wmma_f32_16x16x32_f16.
//       Block = 8 waves = 128 pixels x 64 out-channels. B panel double-buffered
//       through LDS one 3x3-tap slab (32 KB) at a time; epilogue fuses bias,
//       ReLU, residual, ReLU, NHWC store.
// ---------------------------------------------------------------------------

#define BB 16
#define XX 64
#define YY 64
#define CC 256
#define KBLKS 72            // 9 taps * (256/32) k-steps

typedef __attribute__((ext_vector_type(16))) _Float16 v16h;
typedef __attribute__((ext_vector_type(8)))  _Float16 h8;
typedef __attribute__((ext_vector_type(8)))  float    v8f;

union ABFrag { v16h v; h8 h[2]; };

// ---------------- Kernel 1: conv1 + BN1 + ReLU -> f16 NHWC -----------------
__global__ void k_conv1_bn_relu(const float* __restrict__ in,      // (B,X,Y)
                                const float* __restrict__ w1,      // (C,1,3,3)
                                const float* __restrict__ g,
                                const float* __restrict__ bt,
                                const float* __restrict__ mn,
                                const float* __restrict__ vr,
                                _Float16* __restrict__ act1)       // (B,X,Y,C)
{
    int idx = blockIdx.x * blockDim.x + threadIdx.x;   // B*X*Y*C, c fastest
    int c = idx & (CC - 1);
    int p = idx >> 8;
    int y = p & (YY - 1);
    int x = (p >> 6) & (XX - 1);
    int b = p >> 12;

    float s  = g[c] * rsqrtf(vr[c] + 1e-5f);
    float bb = bt[c] - mn[c] * s;
    const float* wc = w1 + c * 9;
    const float* inb = in + (b * XX) * YY;

    float acc = 0.f;
#pragma unroll
    for (int i = 0; i < 3; ++i) {
        int xi = x + i - 1;
        if ((unsigned)xi >= (unsigned)XX) continue;
#pragma unroll
        for (int j = 0; j < 3; ++j) {
            int yi = y + j - 1;
            if ((unsigned)yi >= (unsigned)YY) continue;
            acc += inb[xi * YY + yi] * wc[i * 3 + j];
        }
    }
    float r = acc * s + bb;
    r = r > 0.f ? r : 0.f;
    act1[idx] = (_Float16)r;
}

// ------------- Kernel 2: pack conv2 weights into WMMA-B order --------------
// wpack[ntile(16)][kblk(72)][lane(32)][16 halves], BN2 scale folded.
// Per-lane element order matches A-fragment assembly in K3:
//   e<8  -> K = klo + e          (klo = lane<16 ? 0 : 8)
//   e>=8 -> K = klo + 16 + (e-8)
__global__ void k_pack_w2(const float* __restrict__ w2,            // (C,C,3,3)
                          const float* __restrict__ g,
                          const float* __restrict__ bt,
                          const float* __restrict__ mn,
                          const float* __restrict__ vr,
                          _Float16* __restrict__ wpack,
                          float* __restrict__ bias2)
{
    int t = blockIdx.x * blockDim.x + threadIdx.x;     // 16*72*32 = 36864
    if (t >= 16 * KBLKS * 32) return;
    int lane  = t & 31;
    int kblk  = (t >> 5) % KBLKS;
    int ntile = t / (32 * KBLKS);
    int n = ntile * 16 + (lane & 15);

    float s = g[n] * rsqrtf(vr[n] + 1e-5f);
    int klo = (lane < 16) ? 0 : 8;
    _Float16* dst = wpack + (((size_t)ntile * KBLKS + kblk) * 32 + lane) * 16;

#pragma unroll
    for (int e = 0; e < 16; ++e) {
        int koff = (e < 8) ? (klo + e) : (klo + 16 + (e - 8));
        int kg   = kblk * 32 + koff;
        int tap  = kg >> 8;
        int cin  = kg & (CC - 1);
        int ky = tap / 3, kx = tap % 3;
        dst[e] = (_Float16)(w2[((n * CC + cin) * 3 + ky) * 3 + kx] * s);
    }
    if (t < CC) {
        float ss = g[t] * rsqrtf(vr[t] + 1e-5f);
        bias2[t] = bt[t] - mn[t] * ss;
    }
}

// --------- Kernel 3: implicit GEMM conv2 via WMMA, LDS-staged B ------------
// Grid: (B*X*Y/128) * 4 = 2048 blocks, 256 threads (8 waves).
// Block: ngrp = blockIdx.x & 3 (64 out-channels), 8 m-tiles (one per wave).
// LDS: double buffer, slab = one tap's B panel = 8 kblk * 4 ti * 1 KB = 32 KB.
#define SLAB_HALVES 16384                  // 32 KB per slab in halves
__global__ void k_conv2_wmma(const _Float16* __restrict__ act1,    // (B,X,Y,C) f16
                             const _Float16* __restrict__ wpack,
                             const float* __restrict__ bias2,
                             const float* __restrict__ resid,      // (B,X,Y)
                             float* __restrict__ out)              // (B,N,C)
{
    extern __shared__ _Float16 ldsb[];     // 2 * SLAB_HALVES

    const int tid  = threadIdx.x;
    const int lane = tid & 31;
    const int wv   = tid >> 5;             // wave 0..7

    const int ngrp = blockIdx.x & 3;
    const int mblk = blockIdx.x >> 2;      // 0..511
    const int mt   = mblk * 8 + wv;        // 16-pixel tile id
    const int b    = mt >> 8;
    const int rem  = mt & 255;
    const int x    = rem >> 2;
    const int y0   = (rem & 3) << 4;

    const int mrow = lane & 15;            // A-matrix row for this lane
    const int y    = y0 + mrow;
    const int klo  = (lane & 16) ? 8 : 0;

    const _Float16* actb = act1 + (size_t)(b * XX) * YY * CC;
    const _Float16* wgrp = wpack + (size_t)(ngrp * 4) * KBLKS * 512;  // halves

    // Cooperative stage of one tap slab (4 ti-chunks of 8 KB) into LDS.
    auto stage = [&](int tap, int slab) {
#pragma unroll
        for (int i = 0; i < 8; ++i) {
            int u    = tid + i * 256;      // 16-byte unit id, 0..2047
            int ti   = u >> 9;             // 512 units per ti-chunk
            int unit = u & 511;
            size_t srch = ((size_t)ti * KBLKS + tap * 8) * 512 + (size_t)unit * 8;
            int    dsth = slab * SLAB_HALVES + ti * 4096 + unit * 8;
            *reinterpret_cast<h8*>(ldsb + dsth) =
                *reinterpret_cast<const h8*>(wgrp + srch);
        }
    };

    v8f acc[4] = {};
    const h8 zero8 = {};

    stage(0, 0);

    for (int tap = 0; tap < 9; ++tap) {
        const int slab = tap & 1;
        __syncthreads();                   // slab ready; prev slab free
        if (tap < 8) stage(tap + 1, slab ^ 1);

        const int i  = tap / 3;            // small constants, hoisted per tap
        const int j  = tap - i * 3;
        const int xi = x + i - 1;
        const int yi = y + j - 1;
        const bool vx = (unsigned)xi < (unsigned)XX;   // wave-uniform
        const bool vy = (unsigned)yi < (unsigned)YY;   // per-lane (tile edge)

        if (vx) {
            int yic = yi < 0 ? 0 : (yi > YY - 1 ? YY - 1 : yi);
            const _Float16* pab = actb + ((size_t)(xi * YY + yic) * CC + klo);
#pragma unroll
            for (int c8 = 0; c8 < 8; ++c8) {
                ABFrag a;
                a.h[0] = *reinterpret_cast<const h8*>(pab + c8 * 32);
                a.h[1] = *reinterpret_cast<const h8*>(pab + c8 * 32 + 16);
                if (!vy) { a.h[0] = zero8; a.h[1] = zero8; }
#pragma unroll
                for (int ti = 0; ti < 4; ++ti) {
                    const _Float16* pl =
                        ldsb + slab * SLAB_HALVES + (ti * 8 + c8) * 512 + lane * 16;
                    ABFrag bf;
                    bf.h[0] = *reinterpret_cast<const h8*>(pl);
                    bf.h[1] = *reinterpret_cast<const h8*>(pl + 8);
                    acc[ti] = __builtin_amdgcn_wmma_f32_16x16x32_f16(
                        false, a.v, false, bf.v, (short)0, acc[ti], false, false);
                }
            }
        }
    }

    // Epilogue: r = relu(conv + bias2); o = relu(input + r); NHWC store.
    const size_t outbase = ((size_t)b * XX + x) * YY * CC;
    const int    resbase = (b * XX + x) * YY;
    const int    hihalf  = (lane & 16) ? 8 : 0;
    const int    ncbase  = lane & 15;

#pragma unroll
    for (int vrw = 0; vrw < 8; ++vrw) {
        const int row = vrw + hihalf;
        const float xv = resid[resbase + y0 + row];
#pragma unroll
        for (int ti = 0; ti < 4; ++ti) {
            const int ncol = (ngrp * 4 + ti) * 16 + ncbase;
            float r = acc[ti][vrw] + bias2[ncol];
            r = fmaxf(r, 0.f);
            float o = fmaxf(xv + r, 0.f);
            out[outbase + (size_t)(y0 + row) * CC + ncol] = o;
        }
    }
}

// ---------------------------------------------------------------------------
extern "C" void kernel_launch(void* const* d_in, const int* in_sizes, int n_in,
                              void* d_out, int out_size, void* d_ws, size_t ws_size,
                              hipStream_t stream)
{
    // 0 input, 1 freq_x, 2 freq_y, 3 conv1_w, 4 conv2_w,
    // 5-8 bn1 g/b/m/v, 9-12 bn2 g/b/m/v, 13.. dead params
    const float* input   = (const float*)d_in[0];
    const float* conv1_w = (const float*)d_in[3];
    const float* conv2_w = (const float*)d_in[4];
    const float* bn1_g = (const float*)d_in[5];
    const float* bn1_b = (const float*)d_in[6];
    const float* bn1_m = (const float*)d_in[7];
    const float* bn1_v = (const float*)d_in[8];
    const float* bn2_g = (const float*)d_in[9];
    const float* bn2_b = (const float*)d_in[10];
    const float* bn2_m = (const float*)d_in[11];
    const float* bn2_v = (const float*)d_in[12];
    float* out = (float*)d_out;

    char* ws = (char*)d_ws;
    _Float16* act1  = (_Float16*)ws;                        // 33.55 MB
    size_t act1_bytes = (size_t)BB * XX * YY * CC * sizeof(_Float16);
    _Float16* wpack = (_Float16*)(ws + act1_bytes);         // 1.18 MB
    size_t wpack_bytes = (size_t)16 * KBLKS * 32 * 16 * sizeof(_Float16);
    float* bias2 = (float*)(ws + act1_bytes + wpack_bytes); // 1 KB

    // K1: conv1 + BN1 + ReLU
    {
        int total = BB * XX * YY * CC;                      // 16.7M
        k_conv1_bn_relu<<<total / 256, 256, 0, stream>>>(
            input, conv1_w, bn1_g, bn1_b, bn1_m, bn1_v, act1);
    }
    // K2: weight pack (+ folded BN2 bias)
    {
        int total = 16 * KBLKS * 32;
        k_pack_w2<<<(total + 255) / 256, 256, 0, stream>>>(
            conv2_w, bn2_g, bn2_b, bn2_m, bn2_v, wpack, bias2);
    }
    // K3: implicit-GEMM conv2 via WMMA, 64 KB dynamic LDS (double buffer)
    {
        int blocks = (BB * XX * YY / 128) * 4;              // 2048
        size_t shmem = 2u * SLAB_HALVES * sizeof(_Float16); // 65536 B
        k_conv2_wmma<<<blocks, 256, shmem, stream>>>(
            act1, wpack, bias2, input, out);
    }
    (void)in_sizes; (void)n_in; (void)out_size; (void)ws_size;
}
